// ScaledDotProductAttention_71588514889823
// MI455X (gfx1250) — matrix operand order repaired and checked by reading, not compiled, not run on hardware
//
#include <hip/hip_runtime.h>

// ---- CDNA5 (gfx1250) ReLU-attention, wave32 + v_wmma_f32_16x16x32_f16 ----
// Memory-bound on the mask stream (537 MB @ 23.3 TB/s ~= 26 us floor); K/V
// (33 MB total) live in the 192 MB L2 and are reused by every Q tile of the
// same head. Matmuls run in f16 WMMA with f32 accumulate.

typedef __attribute__((ext_vector_type(16))) _Float16 v16h;
typedef __attribute__((ext_vector_type(2)))  _Float16 v2h;
typedef __attribute__((ext_vector_type(8)))  float    v8f;
typedef __attribute__((ext_vector_type(4)))  float    vf4;
typedef __attribute__((ext_vector_type(2)))  float    vf2;

#define S_LEN 2048
#define D_DIM 64
#define WAVES_PER_BLOCK 8   // 256 threads = 8 wave32

union FragH { v16h v; v2h h[8]; };

// one v_cvt_pk_rtz_f16_f32 per pair of floats
__device__ __forceinline__ v2h pkh(float lo, float hi) {
  auto r = __builtin_amdgcn_cvt_pkrtz(lo, hi);
  return __builtin_bit_cast(v2h, r);
}

__global__ void __launch_bounds__(256)
relu_attn_kernel(const float* __restrict__ Q, const float* __restrict__ K,
                 const float* __restrict__ V, const int* __restrict__ scale_p,
                 const float* __restrict__ M, float* __restrict__ O)
{
  // per-wave 16x32 f32 staging tile for the P (post-ReLU) cross-lane transpose
  __shared__ float plds[WAVES_PER_BLOCK][16 * 32];

  const int lane = threadIdx.x & 31;
  const int wave = threadIdx.x >> 5;
  const int tile = blockIdx.x * WAVES_PER_BLOCK + wave; // 0..4095
  const int bh   = tile >> 7;          // tile / (S/16)
  const int q0   = (tile & 127) << 4;  // query-row base

  const int row  = lane & 15;          // M (A/C) or N (B) index
  const int hi16 = lane >> 4;          // lane-half selector per ISA layouts

  const size_t plane = (size_t)bh * S_LEN * D_DIM;
  const float* Qb = Q + plane;
  const float* Kb = K + plane;
  const float* Vb = V + plane;
  const float* Mb = M + (size_t)bh * S_LEN * S_LEN;
  float*       Ob = O + plane;

  const float inv_scale = 1.0f / (float)(*scale_p);

  // ---- Q tile -> two 16x32 f16 A-fragments (d = 0..31, 32..63)
  // A as packed here (contiguous k, matching B): lane holds row M=lane&15; element e -> k = hi16*16 + e
  FragH aQ[2];
  {
    const float* qrow = Qb + (size_t)(q0 + row) * D_DIM;
#pragma unroll
    for (int f = 0; f < 2; ++f) {
      const vf4* p0 = (const vf4*)(qrow + f * 32 + hi16 * 16);     // 32B aligned
      const vf4* p1 = (const vf4*)(qrow + f * 32 + hi16 * 16 + 8);
      vf4 a = p0[0], b = p0[1], c = p1[0], d = p1[1];
      aQ[f].h[0] = pkh(a.x, a.y);  aQ[f].h[1] = pkh(a.z, a.w);
      aQ[f].h[2] = pkh(b.x, b.y);  aQ[f].h[3] = pkh(b.z, b.w);
      aQ[f].h[4] = pkh(c.x, c.y);  aQ[f].h[5] = pkh(c.z, c.w);
      aQ[f].h[6] = pkh(d.x, d.y);  aQ[f].h[7] = pkh(d.z, d.w);
    }
  }

  v8f acc[4] = {};          // 16x64 output tile, f32
  float* myLds = plds[wave];

#pragma unroll 1
  for (int kb = 0; kb < S_LEN; kb += 32) {
    // prefetch next iteration's mask tile (the streaming-bound operand)
    if (kb + 32 < S_LEN)
      __builtin_prefetch(Mb + (size_t)(q0 + row + hi16 * 8) * S_LEN + kb + 32, 0, 0);

    // ---- K tile -> B fragments bK[t][f]: B[k=d][n=key]
    // B as packed here (contiguous k, matching A): lane holds col N=lane&15; element e -> k = hi16*16 + e
    FragH bK[2][2];
#pragma unroll
    for (int t = 0; t < 2; ++t) {
      const float* krow = Kb + (size_t)(kb + t * 16 + row) * D_DIM + hi16 * 16;
#pragma unroll
      for (int f = 0; f < 2; ++f) {
        const vf4* p = (const vf4*)(krow + f * 32);                // 64B aligned
        vf4 a = p[0], b = p[1], c = p[2], d = p[3];
        bK[t][f].h[0] = pkh(a.x, a.y);  bK[t][f].h[1] = pkh(a.z, a.w);
        bK[t][f].h[2] = pkh(b.x, b.y);  bK[t][f].h[3] = pkh(b.z, b.w);
        bK[t][f].h[4] = pkh(c.x, c.y);  bK[t][f].h[5] = pkh(c.z, c.w);
        bK[t][f].h[6] = pkh(d.x, d.y);  bK[t][f].h[7] = pkh(d.z, d.w);
      }
    }

    // ---- V tile -> B fragments bV[nb]: B[k=key 0..31][n = nb*16+row]
    FragH bV[4];
#pragma unroll
    for (int nb = 0; nb < 4; ++nb) {
      const float* vcol = Vb + (size_t)(kb + hi16 * 16) * D_DIM + nb * 16 + row;
#pragma unroll
      for (int e = 0; e < 8; ++e)
        bV[nb].h[e] = pkh(vcol[(2 * e) * D_DIM], vcol[(2 * e + 1) * D_DIM]);
    }

    // ---- scores: two 16x16 tiles, K-dim 64 = 2 chained WMMAs each
    v8f s[2];
#pragma unroll
    for (int t = 0; t < 2; ++t) {
      v8f z = {};
      z = __builtin_amdgcn_wmma_f32_16x16x32_f16(false, aQ[0].v, false, bK[t][0].v,
                                                 (short)0, z, false, false);
      z = __builtin_amdgcn_wmma_f32_16x16x32_f16(false, aQ[1].v, false, bK[t][1].v,
                                                 (short)0, z, false, false);
      s[t] = z;
    }

    // ---- P = relu(S * inv_scale * mask); stage f32 16x32 tile in LDS
    // C layout: lane,vgpr r -> m = r + hi16*8, n = lane&15
#pragma unroll
    for (int t = 0; t < 2; ++t) {
#pragma unroll
      for (int r = 0; r < 8; ++r) {
        const int m = r + hi16 * 8;
        float mv = __builtin_nontemporal_load(
            Mb + (size_t)(q0 + m) * S_LEN + kb + t * 16 + row);
        float p = fmaxf(s[t][r] * inv_scale * mv, 0.0f);
        myLds[m * 32 + t * 16 + row] = p;
      }
    }
    asm volatile("s_wait_dscnt 0x0" ::: "memory");

    // ---- re-read P in A-fragment order (cross-lane transpose via LDS)
    FragH aP;
    {
      const vf2* p0 = (const vf2*)(myLds + row * 32 + hi16 * 16);      // k hi16*16 + 0..7
      const vf2* p1 = (const vf2*)(myLds + row * 32 + hi16 * 16 + 8);  // k hi16*16 + 8..15
#pragma unroll
      for (int e = 0; e < 4; ++e) {
        vf2 a = p0[e], b = p1[e];
        aP.h[e]     = pkh(a.x, a.y);
        aP.h[4 + e] = pkh(b.x, b.y);
      }
    }
    asm volatile("s_wait_dscnt 0x0" ::: "memory"); // reads land before next iter's stores

    // ---- O += P x V (K-dim = 32 keys, fully used)
#pragma unroll
    for (int nb = 0; nb < 4; ++nb)
      acc[nb] = __builtin_amdgcn_wmma_f32_16x16x32_f16(false, aP.v, false, bV[nb].v,
                                                       (short)0, acc[nb], false, false);
  }

  // ---- store output tile (f32)
#pragma unroll
  for (int nb = 0; nb < 4; ++nb) {
#pragma unroll
    for (int r = 0; r < 8; ++r) {
      const int m = r + hi16 * 8;
      __builtin_nontemporal_store(acc[nb][r],
                                  Ob + (size_t)(q0 + m) * D_DIM + nb * 16 + row);
    }
  }
}

extern "C" void kernel_launch(void* const* d_in, const int* in_sizes, int n_in,
                              void* d_out, int out_size, void* d_ws, size_t ws_size,
                              hipStream_t stream) {
  (void)in_sizes; (void)n_in; (void)out_size; (void)d_ws; (void)ws_size;
  const float* Q     = (const float*)d_in[0];
  const float* K     = (const float*)d_in[1];
  const float* V     = (const float*)d_in[2];
  const int*   scale = (const int*)  d_in[3];
  const float* mask  = (const float*)d_in[4];
  float*       out   = (float*)d_out;

  // B*H*(S/16) = 4096 query tiles, 8 waves per 256-thread block -> 512 blocks
  dim3 grid(512), block(256);
  hipLaunchKernelGGL(relu_attn_kernel, grid, block, 0, stream,
                     Q, K, V, scale, mask, out);
}
